// GraphVAE_3702261809253
// MI455X (gfx1250) — matrix operand order repaired and checked by reading, not compiled, not run on hardware
//
#include <hip/hip_runtime.h>
#include <cstdint>
#include <cstddef>

// ---------------- problem constants (match reference) ----------------
#define N_NODES 50000
#define N_EDGES 1600000
#define F_IN    128
#define G_GR    512
#define MAXN    128
#define LATD    64

// ---------------- CDNA5 WMMA types ----------------
typedef __attribute__((ext_vector_type(16))) __bf16 v16bf;
typedef __attribute__((ext_vector_type(8)))  float  v8f;

union FragBF {
  v16bf v;
  unsigned short s[16];
  uint4 q[2];
};

__device__ __forceinline__ unsigned short f32_to_bf16(float f) {
  unsigned int u = __builtin_bit_cast(unsigned int, f);
  unsigned int r = (u + 0x7FFFu + ((u >> 16) & 1u)) >> 16;   // RNE
  return (unsigned short)r;
}

__device__ __forceinline__ float fast_sigmoid(float v) {
  // v_exp_f32 + v_add + v_rcp_f32 (avoid IEEE div expansion)
  return __builtin_amdgcn_rcpf(1.0f + __expf(-v));
}

__device__ __forceinline__ float hash_normal(unsigned int idx) {
  // PCG-style hash -> Box-Muller (stands in for jax.random.normal(key 42))
  unsigned int s = idx * 747796405u + 2891336453u;
  unsigned int w = ((s >> ((s >> 28u) + 4u)) ^ s) * 277803737u;
  unsigned int u1 = (w >> 22u) ^ w;
  s = (idx ^ 0x9E3779B9u) * 2654435761u + 1013904223u;
  w = ((s >> ((s >> 28u) + 4u)) ^ s) * 277803737u;
  unsigned int u2 = (w >> 22u) ^ w;
  float a = ((float)u1 + 0.5f) * (1.0f / 4294967296.0f);
  float b = ((float)u2 + 0.5f) * (1.0f / 4294967296.0f);
  return sqrtf(-2.0f * logf(a)) * __cosf(6.28318530717958647f * b);
}

// ---------------- elementwise helpers ----------------
__global__ void k_fill_f32(float* __restrict__ p, float v, int n) {
  int i = blockIdx.x * blockDim.x + threadIdx.x;
  if (i < n) p[i] = v;
}

__global__ void k_cast_f32_bf16(const float* __restrict__ s, unsigned short* __restrict__ d, int n) {
  int i = blockIdx.x * blockDim.x + threadIdx.x;
  if (i < n) d[i] = f32_to_bf16(s[i]);
}

__global__ void k_relu_cast_bf16(const float* __restrict__ s, unsigned short* __restrict__ d, int n) {
  int i = blockIdx.x * blockDim.x + threadIdx.x;
  if (i < n) d[i] = f32_to_bf16(fmaxf(s[i], 0.0f));
}

// ---------------- pack f32 weights (KxN row-major) into WMMA B-fragment layout ----
// frag(kt,nt) is a contiguous 512-ushort block; element j of lane l holds
// B[kt*32 + (l>>4)*16 + j][nt*16 + (l&15)].
__global__ void k_pack_b(const float* __restrict__ W, unsigned short* __restrict__ Bp,
                         int K, int N) {
  int idx = blockIdx.x * blockDim.x + threadIdx.x;
  if (idx >= K * N) return;
  int k = idx / N, n = idx - k * N;
  int kt = k >> 5, kk = k & 31;
  int hsel = kk >> 4, j = kk & 15;
  int nt = n >> 4, l15 = n & 15;
  size_t dst = (((size_t)kt * (N >> 4) + nt) * 32 + (hsel * 16 + l15)) * 16 + j;
  Bp[dst] = f32_to_bf16(W[idx]);
}

// ---------------- graph normalization ----------------
__global__ void k_deg(const long long* __restrict__ col, float* __restrict__ deg, int e) {
  int i = blockIdx.x * blockDim.x + threadIdx.x;
  if (i < e) atomicAdd(&deg[(int)col[i]], 1.0f);
}

__global__ void k_dinv(const float* __restrict__ deg, float* __restrict__ dinv, int n) {
  int i = blockIdx.x * blockDim.x + threadIdx.x;
  if (i < n) { float d = deg[i]; dinv[i] = (d > 0.0f) ? rsqrtf(d) : 0.0f; }
}

// out[i] = b + dinv[i]^2 * Hlin[i]   (self-loop contribution + bias); fo = 1<<fsh
__global__ void k_agg_init(const float* __restrict__ Hlin, const float* __restrict__ dinv,
                           const float* __restrict__ bias, float* __restrict__ out,
                           int n, int fsh) {
  int idx = blockIdx.x * blockDim.x + threadIdx.x;
  if (idx >= (n << fsh)) return;
  int i = idx >> fsh, f = idx & ((1 << fsh) - 1);
  float di = dinv[i];
  out[idx] = bias[f] + di * di * Hlin[idx];
}

// out[col] += dinv[row]*dinv[col] * Hlin[row]; 4 feats/thread; fo = 1<<fsh
__global__ void k_edge_agg(const long long* __restrict__ row, const long long* __restrict__ col,
                           const float* __restrict__ dinv, const float* __restrict__ Hlin,
                           float* __restrict__ out, int e, int fsh) {
  int idx = blockIdx.x * blockDim.x + threadIdx.x;
  const int psh = fsh - 2;                       // per-edge thread count = fo/4
  if (idx >= (e << psh)) return;
  int ei = idx >> psh;
  int f4 = (idx & ((1 << psh) - 1)) << 2;
  int r = (int)row[ei], c = (int)col[ei];
  float nrm = dinv[r] * dinv[c];
  const float4 h = *(const float4*)(Hlin + ((size_t)r << fsh) + f4);
  float* o = out + ((size_t)c << fsh) + f4;
  atomicAdd(o + 0, nrm * h.x);
  atomicAdd(o + 1, nrm * h.y);
  atomicAdd(o + 2, nrm * h.z);
  atomicAdd(o + 3, nrm * h.w);
}

// ---------------- WMMA bf16 GEMM: C = A(MxK) @ B(KxN) + bias ----------------
// A: bf16 row-major (lda=K); Bp: pre-packed fragments; 2 N-tiles per wave.
template <bool OUT_BF16, bool RELU>
__global__ void k_wmma_gemm(const unsigned short* __restrict__ A,
                            const unsigned short* __restrict__ Bp,
                            const float* __restrict__ bias,
                            float* __restrict__ Cf, unsigned short* __restrict__ Cb,
                            int M, int K, int Nout) {
  const int lane = threadIdx.x;                  // 0..31 (wave32)
  const int tileM = blockIdx.x * blockDim.y + threadIdx.y;
  const int tileN0 = blockIdx.y * 2;
  if (tileM * 16 >= M) return;                   // wave-uniform guard: EXEC stays all-1s

  const int Ntiles = Nout >> 4;
  const int l15 = lane & 15;
  const int hsel = lane >> 4;
  const int kseg = hsel * 8;

  const unsigned short* Arow = A + (size_t)(tileM * 16 + l15) * K;

  v8f acc0 = {}, acc1 = {};
  for (int k = 0; k < K; k += 32) {
    if (k + 32 < K) __builtin_prefetch(Arow + k + 32, 0, 1);   // global_prefetch_b8
    FragBF a, b0, b1;
    // A 16x32 frag: row=l15, K in {hsel*16..hsel*16+15}
    a.q[0] = *(const uint4*)(Arow + k + hsel * 16);
    a.q[1] = *(const uint4*)(Arow + k + hsel * 16 + 8);
    // packed B frags: contiguous 32B per lane
    const unsigned short* bp =
        Bp + (((size_t)(k >> 5) * Ntiles + tileN0) * 32 + lane) * 16;
    b0.q[0] = *(const uint4*)(bp + 0);
    b0.q[1] = *(const uint4*)(bp + 8);
    b1.q[0] = *(const uint4*)(bp + 512);
    b1.q[1] = *(const uint4*)(bp + 520);
    acc0 = __builtin_amdgcn_wmma_f32_16x16x32_bf16(false, a.v, false, b0.v,
                                                   (short)0, acc0, false, false);
    acc1 = __builtin_amdgcn_wmma_f32_16x16x32_bf16(false, a.v, false, b1.v,
                                                   (short)0, acc1, false, false);
  }

  const int nc0 = tileN0 * 16 + l15;
  const int nc1 = nc0 + 16;
  const float bv0 = bias ? bias[nc0] : 0.0f;
  const float bv1 = bias ? bias[nc1] : 0.0f;
  const int mbase = tileM * 16 + hsel * 8;
#pragma unroll
  for (int r = 0; r < 8; ++r) {
    float v0 = acc0[r] + bv0;
    float v1 = acc1[r] + bv1;
    if (RELU) { v0 = fmaxf(v0, 0.0f); v1 = fmaxf(v1, 0.0f); }
    size_t off = (size_t)(mbase + r) * Nout;
    if (OUT_BF16) {
      Cb[off + nc0] = f32_to_bf16(v0);
      Cb[off + nc1] = f32_to_bf16(v1);
    } else {
      Cf[off + nc0] = v0;
      Cf[off + nc1] = v1;
    }
  }
}

// ---------------- pooling ----------------
__global__ void k_pool(const float* __restrict__ h, const long long* __restrict__ batch,
                       float* __restrict__ sums, float* __restrict__ cnts, int n) {
  int idx = blockIdx.x * blockDim.x + threadIdx.x;
  if (idx >= n * 32) return;                       // 128 feats / 4 per thread
  int i = idx >> 5;
  int f = (idx & 31) << 2;
  int g = (int)batch[i];
  const float4 v = *(const float4*)(h + (size_t)i * 128 + f);
  float* sg = sums + (size_t)g * 128 + f;
  atomicAdd(sg + 0, fmaxf(v.x, 0.0f));             // relu fused into pool read
  atomicAdd(sg + 1, fmaxf(v.y, 0.0f));
  atomicAdd(sg + 2, fmaxf(v.z, 0.0f));
  atomicAdd(sg + 3, fmaxf(v.w, 0.0f));
  if (f == 0) atomicAdd(&cnts[g], 1.0f);
}

__global__ void k_pooled(const float* __restrict__ sums, const float* __restrict__ cnts,
                         float* __restrict__ pooled) {
  int idx = blockIdx.x * blockDim.x + threadIdx.x;
  if (idx >= G_GR * 128) return;
  int g = idx >> 7;
  pooled[idx] = sums[idx] / fmaxf(cnts[g], 1.0f);
}

// ---------------- mu / logvar / reparameterize ----------------
__global__ void k_mulvz(const float* __restrict__ pooled,
                        const float* __restrict__ Wmu, const float* __restrict__ bmu,
                        const float* __restrict__ Wlv, const float* __restrict__ blv,
                        float* __restrict__ mu_out, float* __restrict__ lv_out,
                        float* __restrict__ z) {
  int idx = blockIdx.x * blockDim.x + threadIdx.x;
  if (idx >= G_GR * LATD) return;
  int g = idx >> 6, j = idx & 63;
  const float* pr = pooled + (size_t)g * 128;
  float m = bmu[j], l = blv[j];
  for (int k = 0; k < 128; ++k) {
    float p = pr[k];
    m = fmaf(p, Wmu[(size_t)k * LATD + j], m);
    l = fmaf(p, Wlv[(size_t)k * LATD + j], l);
  }
  mu_out[idx] = m;
  lv_out[idx] = l;
  z[idx] = m + hash_normal(idx) * __expf(0.5f * l);
}

// ---------------- decoder dense1 + BN ----------------
__global__ void k_dense1(const float* __restrict__ z, const float* __restrict__ Wd1,
                         const float* __restrict__ bd1, float* __restrict__ h1) {
  int idx = blockIdx.x * blockDim.x + threadIdx.x;
  if (idx >= G_GR * 256) return;
  int g = idx >> 8, j = idx & 255;
  const float* zr = z + (size_t)g * LATD;
  float a = bd1[j];
  for (int k = 0; k < LATD; ++k) a = fmaf(zr[k], Wd1[(size_t)k * 256 + j], a);
  h1[idx] = fmaxf(a, 0.0f);
}

__global__ void k_bn_stats(const float* __restrict__ h1, float* __restrict__ mean,
                           float* __restrict__ var) {
  __shared__ float ss[128], sq[128];
  int j = blockIdx.x, t = threadIdx.x;
  float s = 0.0f, q = 0.0f;
  for (int r = t; r < G_GR; r += 128) {
    float v = h1[(size_t)r * 256 + j];
    s += v; q += v * v;
  }
  ss[t] = s; sq[t] = q;
  __syncthreads();
  for (int o = 64; o > 0; o >>= 1) {
    if (t < o) { ss[t] += ss[t + o]; sq[t] += sq[t + o]; }
    __syncthreads();
  }
  if (t == 0) {
    float m = ss[0] * (1.0f / G_GR);
    mean[j] = m;
    var[j] = sq[0] * (1.0f / G_GR) - m * m;
  }
}

__global__ void k_bn_apply(const float* __restrict__ h1, const float* __restrict__ mean,
                           const float* __restrict__ var, const float* __restrict__ gamma,
                           const float* __restrict__ beta, unsigned short* __restrict__ h1bf) {
  int idx = blockIdx.x * blockDim.x + threadIdx.x;
  if (idx >= G_GR * 256) return;
  int j = idx & 255;
  float v = gamma[j] * (h1[idx] - mean[j]) * rsqrtf(var[j] + 1e-5f) + beta[j];
  h1bf[idx] = f32_to_bf16(fmaxf(v, 0.0f));
}

// ---------------- adj = sigmoid(R @ R^T, diag->0) per graph, WMMA K=32 ----------------
__global__ void k_adj(const unsigned short* __restrict__ R, float* __restrict__ out) {
  const int lane = threadIdx.x;
  const int g = blockIdx.z;
  const int tileM = blockIdx.x, tileN = blockIdx.y;
  const int l15 = lane & 15;
  const int hsel = lane >> 4;
  const unsigned short* Rg = R + (size_t)g * (MAXN * 32);

  FragBF a, b;
  const unsigned short* ra = Rg + (size_t)(tileM * 16 + l15) * 32;
  const unsigned short* rb = Rg + (size_t)(tileN * 16 + l15) * 32;
  a.q[0] = *(const uint4*)(ra + hsel * 8);          // A 16x32 frag
  a.q[1] = *(const uint4*)(ra + 16 + hsel * 8);
  b.q[0] = *(const uint4*)(rb + hsel * 8);          // B = R^T 32x16 frag (rows of R)
  b.q[1] = *(const uint4*)(rb + 16 + hsel * 8);

  v8f acc = {};
  acc = __builtin_amdgcn_wmma_f32_16x16x32_bf16(false, a.v, false, b.v,
                                                (short)0, acc, false, false);

  float* og = out + (size_t)g * (MAXN * MAXN);
  const int nc = tileN * 16 + l15;
  const int mb = tileM * 16 + hsel * 8;
#pragma unroll
  for (int r = 0; r < 8; ++r) {
    int mr = mb + r;
    float v = (mr == nc) ? 0.0f : acc[r];           // adj - diag*I  =>  sigmoid(0)=0.5 on diag
    og[(size_t)mr * MAXN + nc] = fast_sigmoid(v);   // exp + add + v_rcp_f32
  }
}

// ---------------- host-side orchestration ----------------
static inline int cdiv(int a, int b) { return (a + b - 1) / b; }

extern "C" void kernel_launch(void* const* d_in, const int* in_sizes, int n_in,
                              void* d_out, int out_size, void* d_ws, size_t ws_size,
                              hipStream_t stream) {
  (void)in_sizes; (void)n_in; (void)out_size; (void)ws_size;

  const float*     x     = (const float*)d_in[0];
  const long long* ei    = (const long long*)d_in[1];
  const long long* batch = (const long long*)d_in[2];
  const float *W1 = (const float*)d_in[3],  *b1 = (const float*)d_in[4];
  const float *W2 = (const float*)d_in[5],  *b2 = (const float*)d_in[6];
  const float *W3 = (const float*)d_in[7],  *b3 = (const float*)d_in[8];
  const float *Wmu = (const float*)d_in[9],  *bmu = (const float*)d_in[10];
  const float *Wlv = (const float*)d_in[11], *blv = (const float*)d_in[12];
  const float *Wd1 = (const float*)d_in[13], *bd1 = (const float*)d_in[14];
  const float *gamma = (const float*)d_in[15], *beta = (const float*)d_in[16];
  const float *Wd2 = (const float*)d_in[17], *bd2 = (const float*)d_in[18];

  const long long* rowi = ei;            // edge_index[0] : source j
  const long long* coli = ei + N_EDGES;  // edge_index[1] : target i

  // ---- workspace carve ----
  char* p = (char*)d_ws;
  auto carve = [&](size_t bytes) -> void* {
    void* r = (void*)p;
    p += (bytes + 255) & ~(size_t)255;
    return r;
  };
  float*          deg    = (float*)carve((size_t)N_NODES * 4);
  float*          dinv   = (float*)carve((size_t)N_NODES * 4);
  unsigned short* Xbf    = (unsigned short*)carve((size_t)N_NODES * 128 * 2);
  float*          Hlin   = (float*)carve((size_t)N_NODES * 128 * 4);
  float*          Hagg   = (float*)carve((size_t)N_NODES * 128 * 4);
  unsigned short* Bpack  = (unsigned short*)carve((size_t)256 * 4096 * 2); // reused per GEMM
  float*          sums   = (float*)carve((size_t)G_GR * 128 * 4);
  float*          cnts   = (float*)carve((size_t)G_GR * 4);
  float*          pooled = (float*)carve((size_t)G_GR * 128 * 4);
  float*          zbuf   = (float*)carve((size_t)G_GR * LATD * 4);
  float*          h1     = (float*)carve((size_t)G_GR * 256 * 4);
  float*          bnm    = (float*)carve((size_t)256 * 4);
  float*          bnv    = (float*)carve((size_t)256 * 4);
  unsigned short* h1bf   = (unsigned short*)carve((size_t)G_GR * 256 * 2);
  unsigned short* Rbf    = (unsigned short*)carve((size_t)G_GR * MAXN * 32 * 2);

  float* out_adj = (float*)d_out;                                   // G*MAXN*MAXN
  float* out_mu  = out_adj + (size_t)G_GR * MAXN * MAXN;            // G*LAT
  float* out_lv  = out_mu + (size_t)G_GR * LATD;                    // G*LAT

  const int TB = 256;

  // ---- degree / norm (self-loops: deg starts at 1, dinv^2 handled in agg_init) ----
  k_fill_f32<<<cdiv(N_NODES, TB), TB, 0, stream>>>(deg, 1.0f, N_NODES);
  k_deg<<<cdiv(N_EDGES, TB), TB, 0, stream>>>(coli, deg, N_EDGES);
  k_dinv<<<cdiv(N_NODES, TB), TB, 0, stream>>>(deg, dinv, N_NODES);

  // ---- input features to bf16 ----
  k_cast_f32_bf16<<<cdiv(N_NODES * F_IN, TB), TB, 0, stream>>>(x, Xbf, N_NODES * F_IN);

  // ---- 3 GCN layers ----
  const float* Ws[3] = {W1, W2, W3};
  const float* bs[3] = {b1, b2, b3};
  const int fis[3] = {128, 32, 64};
  const int fos[3] = {32, 64, 128};
  const int fsh[3] = {5, 6, 7};                   // log2(fo)
  for (int L = 0; L < 3; ++L) {
    const int fi = fis[L], fo = fos[L];
    k_pack_b<<<cdiv(fi * fo, TB), TB, 0, stream>>>(Ws[L], Bpack, fi, fo);
    dim3 gblk(32, 4);
    dim3 ggrd(cdiv(N_NODES / 16, 4), fo / 32);
    k_wmma_gemm<false, false><<<ggrd, gblk, 0, stream>>>(Xbf, Bpack, nullptr,
                                                         Hlin, nullptr,
                                                         N_NODES, fi, fo);
    k_agg_init<<<cdiv(N_NODES * fo, TB), TB, 0, stream>>>(Hlin, dinv, bs[L], Hagg,
                                                          N_NODES, fsh[L]);
    k_edge_agg<<<cdiv(N_EDGES * (fo / 4), TB), TB, 0, stream>>>(rowi, coli, dinv, Hlin,
                                                                Hagg, N_EDGES, fsh[L]);
    if (L < 2)
      k_relu_cast_bf16<<<cdiv(N_NODES * fo, TB), TB, 0, stream>>>(Hagg, Xbf, N_NODES * fo);
  }

  // ---- global mean pool (relu fused) ----
  k_fill_f32<<<cdiv(G_GR * 128, TB), TB, 0, stream>>>(sums, 0.0f, G_GR * 128);
  k_fill_f32<<<cdiv(G_GR, TB), TB, 0, stream>>>(cnts, 0.0f, G_GR);
  k_pool<<<cdiv(N_NODES * 32, TB), TB, 0, stream>>>(Hagg, batch, sums, cnts, N_NODES);
  k_pooled<<<cdiv(G_GR * 128, TB), TB, 0, stream>>>(sums, cnts, pooled);

  // ---- mu / logvar / z ----
  k_mulvz<<<cdiv(G_GR * LATD, TB), TB, 0, stream>>>(pooled, Wmu, bmu, Wlv, blv,
                                                    out_mu, out_lv, zbuf);

  // ---- decoder dense1 + BN ----
  k_dense1<<<cdiv(G_GR * 256, TB), TB, 0, stream>>>(zbuf, Wd1, bd1, h1);
  k_bn_stats<<<256, 128, 0, stream>>>(h1, bnm, bnv);
  k_bn_apply<<<cdiv(G_GR * 256, TB), TB, 0, stream>>>(h1, bnm, bnv, gamma, beta, h1bf);

  // ---- node_repr = h1 @ Wd2 + bd2 (WMMA, bf16 out) ----
  k_pack_b<<<cdiv(256 * 4096, TB), TB, 0, stream>>>(Wd2, Bpack, 256, 4096);
  {
    dim3 gblk(32, 4);
    dim3 ggrd(cdiv(G_GR / 16, 4), 4096 / 32);
    k_wmma_gemm<true, false><<<ggrd, gblk, 0, stream>>>(h1bf, Bpack, bd2,
                                                        nullptr, Rbf,
                                                        G_GR, 256, 4096);
  }

  // ---- adj = sigmoid(R R^T with zeroed diag), one WMMA per 16x16 tile ----
  {
    dim3 ablk(32, 1, 1);
    dim3 agrd(MAXN / 16, MAXN / 16, G_GR);
    k_adj<<<agrd, ablk, 0, stream>>>(Rbf, out_adj);
  }
}
